// MAGRU_53687091200306
// MI455X (gfx1250) — matrix operand order repaired and checked by reading, not compiled, not run on hardware
//
#include <hip/hip_runtime.h>
#include <hip/hip_bf16.h>

// MAGRU forward for MI455X (gfx1250, wave32, WMMA + TDM).
// Shapes: N=64, T=512, D=128, H=128, A=6, G=3H=384.
//
//  k_prep_w    : w [A][128][384] f32 -> wT [A][384][128] bf16
//  k_gx        : gx[t][g][n] = x[n,t,:] . w_i[a[n,t]][:,g] + b[a[n,t]][g]  (WMMA bf16)
//  k_recurrent : 512-step GRU scan; h in LDS; zr/cand GEMMs via WMMA; the
//                per-step gx slab (16 rows x 384 cols, 24KB) is double-buffered
//                into LDS by the Tensor Data Mover (tensor_load_to_lds +
//                s_wait_tensorcnt) so global latency is off the serial chain.

#define N_ 64
#define T_ 512
#define D_ 128
#define H_ 128
#define A_ 6
#define G_ 384  // 3*H

typedef __attribute__((ext_vector_type(16))) __bf16 v16bf;
typedef __attribute__((ext_vector_type(8)))  __bf16 v8bf;
typedef __attribute__((ext_vector_type(8)))  float  v8f;
typedef __attribute__((ext_vector_type(4)))  float  v4f;
typedef __attribute__((ext_vector_type(4)))  unsigned int v4u;
typedef __attribute__((ext_vector_type(8)))  int    v8i;
typedef __attribute__((ext_vector_type(4)))  int    v4i;

#if __has_builtin(__builtin_amdgcn_tensor_load_to_lds)
#define HAVE_TDM 1
#else
#define HAVE_TDM 0
#endif

static __device__ __forceinline__ v16bf cat8(v8bf lo, v8bf hi) {
  return __builtin_shufflevector(lo, hi, 0,1,2,3,4,5,6,7,8,9,10,11,12,13,14,15);
}

// ---------------------------------------------------------------------------
// Transpose + convert: w [A][128][384] f32 -> wT [A][384][128] bf16
// ---------------------------------------------------------------------------
__global__ __launch_bounds__(256)
void k_prep_w(const float* __restrict__ w, __bf16* __restrict__ wT) {
  int idx = blockIdx.x * 256 + threadIdx.x;          // over A*128*384
  if (idx >= A_ * 128 * G_) return;
  int e = idx / (128 * G_);
  int r = idx % (128 * G_);
  int k = r / G_;        // inner-dim index (D or H)
  int g = r % G_;        // output column
  wT[(e * G_ + g) * 128 + k] = (__bf16)w[idx];
}

// ---------------------------------------------------------------------------
// gx GEMM: one wave per 16-row tile (rows = flattened n*T+t), loop 24 col tiles.
// Output layout gx[T][G][N] so the serial phase's TDM tile is a clean 2D slab.
// ---------------------------------------------------------------------------
__global__ __launch_bounds__(32)
void k_gx(const float* __restrict__ x, const int* __restrict__ a,
          const float* __restrict__ b, const __bf16* __restrict__ wiT,
          float* __restrict__ gx) {
  const int rt   = blockIdx.x;      // 0..2047 (32768 rows / 16)
  const int lane = threadIdx.x;     // 0..31
  const int mrow = lane & 15;
  const int hi   = lane >> 4;

  __shared__ int a_s[16];
  if (lane < 16) a_s[lane] = a[rt * 16 + lane];
  __syncthreads();

  unsigned amask = 0;
  #pragma unroll
  for (int i = 0; i < 16; ++i) amask |= 1u << a_s[i];

  // A fragments: 16x32 bf16 per k-tile; lane holds row M=mrow,
  // K runs [K0,K0+8) and [K0+16,K0+24), K0 = kt*32 + (hi?8:0).
  v16bf aF[4];
  {
    const float* xr = x + (rt * 16 + mrow) * D_;
    #pragma unroll
    for (int kt = 0; kt < 4; ++kt) {
      const int K0 = kt * 32 + (hi ? 8 : 0);
      v4f f0 = *(const v4f*)(xr + K0);
      v4f f1 = *(const v4f*)(xr + K0 + 4);
      v4f f2 = *(const v4f*)(xr + K0 + 16);
      v4f f3 = *(const v4f*)(xr + K0 + 20);
      v16bf af;
      #pragma unroll
      for (int i = 0; i < 4; ++i) {
        af[i]      = (__bf16)f0[i];
        af[4 + i]  = (__bf16)f1[i];
        af[8 + i]  = (__bf16)f2[i];
        af[12 + i] = (__bf16)f3[i];
      }
      aF[kt] = af;
    }
  }

  const int n  = rt >> 5;            // T/16 = 32 tiles per batch row
  const int t0 = (rt & 31) * 16;

  for (int ct = 0; ct < 24; ++ct) {
    const int col = ct * 16 + mrow;
    v8f out;
    #pragma unroll
    for (int j = 0; j < 8; ++j) out[j] = 0.0f;

    for (int e = 0; e < A_; ++e) {           // uniform branch: EXEC stays full
      if (!(amask & (1u << e))) continue;
      v8f acc = {0.f,0.f,0.f,0.f,0.f,0.f,0.f,0.f};
      #pragma unroll
      for (int kt = 0; kt < 4; ++kt) {
        const __bf16* wp = wiT + (e * G_ + col) * D_ + kt * 32 + (hi ? 8 : 0);
        v16bf bF = cat8(*(const v8bf*)wp, *(const v8bf*)(wp + 16));
        acc = __builtin_amdgcn_wmma_f32_16x16x32_bf16(
                  false, aF[kt], false, bF, (short)0, acc, false, false);
      }
      const float bias = b[e * G_ + col];
      #pragma unroll
      for (int j = 0; j < 8; ++j) {
        const int m = j + 8 * hi;
        if (a_s[m] == e) out[j] = acc[j] + bias;
      }
    }
    #pragma unroll
    for (int j = 0; j < 8; ++j) {
      const int m = j + 8 * hi;
      gx[((t0 + m) * G_ + col) * N_ + n] = out[j];
    }
  }
}

// ---------------------------------------------------------------------------
// TDM: async-load one step's gx slab (2D tile: 16 x 384 f32, row width N=64)
// from global into LDS. Issued by one wave; tracked by TENSORcnt.
// ---------------------------------------------------------------------------
static __device__ __forceinline__ void tdm_load_gx_slab(const float* gbase,
                                                        unsigned lds_off) {
#if HAVE_TDM
  const unsigned long long ga = (unsigned long long)(uintptr_t)gbase;
  v4u g0;
  g0[0] = 1u;                                            // count=1, user D#
  g0[1] = lds_off;                                       // lds_addr (bytes)
  g0[2] = (unsigned)(ga & 0xFFFFFFFFu);                  // global_addr[31:0]
  g0[3] = (unsigned)((ga >> 32) & 0x1FFFFFFu)            // global_addr[56:32]
        | (2u << 30);                                    // type=2 ("image")
  v8i g1;
  g1[0] = (int)(2u << 16);     // workgroup_mask=0, data_size=2 (4B), no flags
  g1[1] = (int)(64u << 16);    // tensor_dim0 = 64 (row width N)
  g1[2] = (int)(384u << 16);   // tensor_dim0 hi=0 | tensor_dim1 = 384
  g1[3] = (int)(16u << 16);    // tensor_dim1 hi=0 | tile_dim0 = 16
  g1[4] = (int)384;            // tile_dim1 = 384, tile_dim2 = 0
  g1[5] = (int)64;             // tensor_dim0_stride = 64
  g1[6] = 0;                   // stride hi | tensor_dim1_stride lo (unused, 2D)
  g1[7] = 0;
  const v4i z4 = {0, 0, 0, 0};                 // groups 2/3 unused (2D tensor)
  const v8i z8 = {0, 0, 0, 0, 0, 0, 0, 0};     // extra group (clang-23 form)
  __builtin_amdgcn_tensor_load_to_lds(g0, g1, z4, z4, z8, 0);
#else
  (void)gbase; (void)lds_off;
#endif
}

// ---------------------------------------------------------------------------
// Recurrent scan. grid=4 (16 batch rows each), block=512 (16 waves).
// Stage A: 16 waves cover 16 zr col-tiles (256 cols). Stage B: waves 0..7
// cover 8 candidate col-tiles. h kept in LDS (f32 + bf16 mirror); gx slab
// double-buffered in LDS via TDM.
// ---------------------------------------------------------------------------
__global__ __launch_bounds__(512)
void k_recurrent(const int* __restrict__ a, const unsigned char* __restrict__ reset,
                 const float* __restrict__ carry, const float* __restrict__ init_h,
                 const __bf16* __restrict__ whT, const float* __restrict__ gx,
                 float* __restrict__ out) {
  const int n0   = blockIdx.x * 16;
  const int tid  = threadIdx.x;
  const int lane = tid & 31;
  const int wave = tid >> 5;     // 0..15
  const int mrow = lane & 15;
  const int hi   = lane >> 4;

  __shared__ float         h_f[16][H_];          // h (fp32)
  __shared__ __bf16        h_b[16][H_];          // h (bf16, A-fragment source)
  __shared__ __bf16        rh_b[16][H_];         // r*h (bf16)
  __shared__ float         z_s[16][H_];          // z gate
  __shared__ int           a_s[16];
  __shared__ unsigned char r_s[16];
  __shared__ __align__(16) float gx_s[2][G_ * 16];  // TDM double buffer (48KB)

  const unsigned lds_gx0 = (unsigned)(unsigned long long)(uintptr_t)&gx_s[0][0];
  const unsigned lds_gx1 = (unsigned)(unsigned long long)(uintptr_t)&gx_s[1][0];

#if HAVE_TDM
  if (wave == 0)   // preload step 0 slab into buffer 0
    tdm_load_gx_slab(gx + 0 * (long)G_ * N_ + n0, lds_gx0);
#endif

  for (int t = 0; t < T_; ++t) {
    const int buf = t & 1;

#if HAVE_TDM
    if (wave == 0) {
      if (t + 1 < T_) {   // pipeline: issue t+1, wait for t (<=1 outstanding)
        tdm_load_gx_slab(gx + (long)(t + 1) * G_ * N_ + n0,
                         (t + 1) & 1 ? lds_gx1 : lds_gx0);
        __builtin_amdgcn_s_wait_tensorcnt(1);
      } else {
        __builtin_amdgcn_s_wait_tensorcnt(0);
      }
    }
#else
    (void)lds_gx0; (void)lds_gx1;
    {   // fallback: cooperative copy of slab t
      const float* src = gx + (long)t * G_ * N_ + n0;
      for (int idx = tid; idx < G_ * 4; idx += 512) {
        int g = idx >> 2, q = idx & 3;
        *(v4f*)&gx_s[buf][g * 16 + q * 4] = *(const v4f*)(src + g * N_ + q * 4);
      }
    }
#endif

    if (tid < 16) {
      a_s[tid] = a[(n0 + tid) * T_ + t];
      r_s[tid] = reset[(n0 + tid) * T_ + t];
    }
    __syncthreads();

    // h_pre: reset -> initial_h (post-substituted carry == initial_h on reset);
    // t==0 additionally seeds from carry[:,0].
    for (int idx = tid; idx < 16 * H_; idx += 512) {
      const int m  = idx >> 7;
      const int hh = idx & (H_ - 1);
      float hv;
      if (r_s[m])      hv = init_h[hh];
      else if (t == 0) hv = carry[((long)(n0 + m) * T_) * H_ + hh];
      else             hv = h_f[m][hh];
      h_f[m][hh] = hv;
      h_b[m][hh] = (__bf16)hv;
    }
    __syncthreads();

    unsigned amask = 0;
    #pragma unroll
    for (int i = 0; i < 16; ++i) amask |= 1u << a_s[i];

    // ---- stage A: zr = sigmoid(gx_zr + h @ W_zr[a]) ----
    {
      const int col = wave * 16 + mrow;            // 0..255
      const float* gp = &gx_s[buf][col * 16 + 8 * hi];
      v4f g0 = *(const v4f*)gp;
      v4f g1 = *(const v4f*)(gp + 4);

      v16bf aF[4];
      #pragma unroll
      for (int kt = 0; kt < 4; ++kt) {
        const int K0 = kt * 32 + (hi ? 8 : 0);
        aF[kt] = cat8(*(const v8bf*)&h_b[mrow][K0], *(const v8bf*)&h_b[mrow][K0 + 16]);
      }
      v8f sel;
      #pragma unroll
      for (int j = 0; j < 8; ++j) sel[j] = (j < 4) ? g0[j] : g1[j - 4];

      for (int e = 0; e < A_; ++e) {               // uniform: EXEC stays full
        if (!(amask & (1u << e))) continue;
        v8f acc = {0.f,0.f,0.f,0.f,0.f,0.f,0.f,0.f};
        #pragma unroll
        for (int kt = 0; kt < 4; ++kt) {
          const __bf16* wp = whT + (e * G_ + col) * H_ + kt * 32 + (hi ? 8 : 0);
          v16bf bF = cat8(*(const v8bf*)wp, *(const v8bf*)(wp + 16));
          acc = __builtin_amdgcn_wmma_f32_16x16x32_bf16(
                    false, aF[kt], false, bF, (short)0, acc, false, false);
        }
        #pragma unroll
        for (int j = 0; j < 8; ++j) {
          const int m = j + 8 * hi;
          if (a_s[m] == e) sel[j] = ((j < 4) ? g0[j] : g1[j - 4]) + acc[j];
        }
      }
      #pragma unroll
      for (int j = 0; j < 8; ++j) {
        const int m = j + 8 * hi;
        const float s = 1.0f / (1.0f + __expf(-sel[j]));
        if (wave < 8) z_s[m][col] = s;                                    // z
        else          rh_b[m][col - H_] = (__bf16)(s * h_f[m][col - H_]); // r*h
      }
    }
    __syncthreads();

    // ---- stage B: cand = tanh(gx_a + (r*h) @ W_a[a]); h = (1-z)h + z*cand ----
    if (wave < 8) {
      const int hh0 = wave * 16 + mrow;            // 0..127
      const int col = 2 * H_ + hh0;                // 256..383
      const float* gp = &gx_s[buf][col * 16 + 8 * hi];
      v4f g0 = *(const v4f*)gp;
      v4f g1 = *(const v4f*)(gp + 4);

      v16bf aF[4];
      #pragma unroll
      for (int kt = 0; kt < 4; ++kt) {
        const int K0 = kt * 32 + (hi ? 8 : 0);
        aF[kt] = cat8(*(const v8bf*)&rh_b[mrow][K0], *(const v8bf*)&rh_b[mrow][K0 + 16]);
      }
      v8f sel;
      #pragma unroll
      for (int j = 0; j < 8; ++j) sel[j] = (j < 4) ? g0[j] : g1[j - 4];

      for (int e = 0; e < A_; ++e) {
        if (!(amask & (1u << e))) continue;
        v8f acc = {0.f,0.f,0.f,0.f,0.f,0.f,0.f,0.f};
        #pragma unroll
        for (int kt = 0; kt < 4; ++kt) {
          const __bf16* wp = whT + (e * G_ + col) * H_ + kt * 32 + (hi ? 8 : 0);
          v16bf bF = cat8(*(const v8bf*)wp, *(const v8bf*)(wp + 16));
          acc = __builtin_amdgcn_wmma_f32_16x16x32_bf16(
                    false, aF[kt], false, bF, (short)0, acc, false, false);
        }
        #pragma unroll
        for (int j = 0; j < 8; ++j) {
          const int m = j + 8 * hi;
          if (a_s[m] == e) sel[j] = ((j < 4) ? g0[j] : g1[j - 4]) + acc[j];
        }
      }
      #pragma unroll
      for (int j = 0; j < 8; ++j) {
        const int m = j + 8 * hi;
        const float cand = tanhf(sel[j]);
        const float z    = z_s[m][hh0];
        const float hp   = h_f[m][hh0];
        const float hn   = (1.0f - z) * hp + z * cand;
        h_f[m][hh0] = hn;
        h_b[m][hh0] = (__bf16)hn;
        const long oidx = ((long)(n0 + m) * T_ + t) * H_ + hh0;
        out[oidx] = hn;                                  // outputs_seq
        out[(long)N_ * T_ * H_ + oidx] = hn;             // states_seq (same)
      }
    }
    __syncthreads();
  }

  if (blockIdx.x == 0 && tid < H_)
    out[2L * N_ * T_ * H_ + tid] = init_h[tid];          // init_state [1,H]
}

// ---------------------------------------------------------------------------
extern "C" void kernel_launch(void* const* d_in, const int* in_sizes, int n_in,
                              void* d_out, int out_size, void* d_ws, size_t ws_size,
                              hipStream_t stream) {
  (void)in_sizes; (void)n_in; (void)out_size; (void)ws_size;
  const float*         x      = (const float*)d_in[0];
  const int*           a      = (const int*)d_in[1];
  const unsigned char* reset  = (const unsigned char*)d_in[2];  // jnp bool_
  const float*         carry  = (const float*)d_in[3];
  const float*         w_i    = (const float*)d_in[4];
  const float*         w_h    = (const float*)d_in[5];
  const float*         b      = (const float*)d_in[6];
  const float*         init_h = (const float*)d_in[7];
  float*               out    = (float*)d_out;

  char*   ws  = (char*)d_ws;
  __bf16* wiT = (__bf16*)ws;                               // A*G*D bf16 = 576 KB
  __bf16* whT = (__bf16*)(ws + (size_t)A_ * G_ * 128 * 2); // A*G*H bf16 = 576 KB
  float*  gx  = (float*)(ws + (size_t)A_ * G_ * 128 * 4);  // T*G*N f32  = 48 MB

  const int prepBlocks = (A_ * 128 * G_ + 255) / 256;      // 1152
  k_prep_w<<<prepBlocks, 256, 0, stream>>>(w_i, wiT);
  k_prep_w<<<prepBlocks, 256, 0, stream>>>(w_h, whT);
  k_gx<<<(N_ * T_) / 16, 32, 0, stream>>>(x, a, b, wiT, gx);
  k_recurrent<<<N_ / 16, 512, 0, stream>>>(a, reset, carry, init_h, whT, gx, out);
}
